// BoundaryGuidedDSTLayer_57964878626932
// MI455X (gfx1250) — matrix operand order repaired and checked by reading, not compiled, not run on hardware
//
#include <hip/hip_runtime.h>
#include <hip/hip_bf16.h>
#include <math.h>

#define Bn 4
#define Tn 2048
#define Cn 512
#define Hn 8
#define HDn 64
#define FFn 2048

typedef __bf16 bf16_t;
typedef __bf16 v16bf __attribute__((ext_vector_type(16)));
typedef float  v8f   __attribute__((ext_vector_type(8)));
typedef unsigned int tdm_v4u __attribute__((ext_vector_type(4)));
typedef int          tdm_v8i __attribute__((ext_vector_type(8)));
typedef int          tdm_v4i __attribute__((ext_vector_type(4)));

union Frag {
    v16bf v;
    float4 f4[2];
};

__device__ __forceinline__ float gelu_exact(float x) {
    return 0.5f * x * (1.0f + erff(x * 0.7071067811865476f));
}

__device__ __forceinline__ v8f wmma_bf16(const Frag& a, const Frag& b, v8f c) {
    return __builtin_amdgcn_wmma_f32_16x16x32_bf16(false, a.v, false, b.v, (short)0, c, false, false);
}

// Issue a 2D TDM tile load: global (row-major, stride_elems between rows, bf16)
// -> LDS (packed tile_w x tile_h, row-major). tile_w*2 bytes must be a multiple of 4.
__device__ __forceinline__ void tdm_load_2d(unsigned lds_off, const bf16_t* gptr,
                                            unsigned tile_w, unsigned tile_h,
                                            unsigned stride_elems) {
    unsigned long long ga = (unsigned long long)(size_t)gptr;
    tdm_v4u g0;
    g0[0] = 1u;                                            // count=1, user mode
    g0[1] = lds_off;                                       // lds_addr
    g0[2] = (unsigned)ga;                                  // global_addr[31:0]
    g0[3] = (unsigned)((ga >> 32) & 0x1FFFFFFull) | (2u << 30); // addr[56:32] | type=2
    unsigned td0 = stride_elems;                           // row length valid to stride
    unsigned td1 = 1u << 20;                               // plenty of rows (tile in-bounds)
    tdm_v8i g1;
    g1[0] = (int)(1u << 16);                               // data_size=1 (2 bytes), mask=0
    g1[1] = (int)((td0 & 0xFFFFu) << 16);                  // tensor_dim0[15:0] @ bits 63:48
    g1[2] = (int)((td0 >> 16) | ((td1 & 0xFFFFu) << 16));  // dim0 hi | dim1 lo
    g1[3] = (int)((td1 >> 16) | (tile_w << 16));           // dim1 hi | tile_dim0
    g1[4] = (int)tile_h;                                   // tile_dim1 | tile_dim2=0
    g1[5] = (int)stride_elems;                             // tensor_dim0_stride[31:0]
    g1[6] = 0;                                             // stride0 hi | stride1 lo
    g1[7] = 0;
    tdm_v4i z4 = {0, 0, 0, 0};
    tdm_v8i z8 = {0, 0, 0, 0, 0, 0, 0, 0};
    __builtin_amdgcn_tensor_load_to_lds(g0, g1, z4, z4, z8, 0);
}

// ---------------- pack kernels ----------------

__global__ void pack_transpose(const float* __restrict__ W, bf16_t* __restrict__ Bt, int N, int K) {
    int idx = blockIdx.x * 256 + threadIdx.x;
    if (idx >= N * K) return;
    int n = idx % N;
    int k = idx / N;
    Bt[idx] = (bf16_t)W[(size_t)n * K + k];
}

__global__ void pack_conv_w(const float* __restrict__ W, bf16_t* __restrict__ Bc) {
    int idx = blockIdx.x * 256 + threadIdx.x;            // idx = (kk*C+ci)*C + co
    if (idx >= 3 * Cn * Cn) return;
    int co = idx & (Cn - 1);
    int r  = idx >> 9;
    int kk = r >> 9;
    int ci = r & (Cn - 1);
    Bc[idx] = (bf16_t)W[((size_t)co * Cn + ci) * 3 + kk];
}

__global__ void pack_xpad(const float* __restrict__ x, bf16_t* __restrict__ xp) {
    int idx = blockIdx.x * 256 + threadIdx.x;
    int total = Bn * (Tn + 2) * Cn;
    if (idx >= total) return;
    int c   = idx & (Cn - 1);
    int rem = idx >> 9;
    int t   = rem % (Tn + 2);
    int b   = rem / (Tn + 2);
    float vv = 0.0f;
    if (t >= 1 && t <= Tn) vv = x[((size_t)b * Tn + (t - 1)) * Cn + c];
    xp[idx] = (bf16_t)vv;
}

// ---------------- layernorm ----------------
__global__ __launch_bounds__(256) void layernorm_k(const float* __restrict__ in,
                                                   const float* __restrict__ g,
                                                   const float* __restrict__ bb,
                                                   bf16_t* __restrict__ outb,
                                                   float* __restrict__ outf) {
    int row  = blockIdx.x;
    int tid  = threadIdx.x;
    int lane = tid & 31, wave = tid >> 5;
    const float* p = in + (size_t)row * Cn;
    float x0 = p[tid], x1 = p[tid + 256];
    float s = x0 + x1, ss = x0 * x0 + x1 * x1;
    #pragma unroll
    for (int m = 16; m >= 1; m >>= 1) {
        s  += __shfl_xor(s, m, 32);
        ss += __shfl_xor(ss, m, 32);
    }
    __shared__ float sh[8][2];
    if (lane == 0) { sh[wave][0] = s; sh[wave][1] = ss; }
    __syncthreads();
    float ts = 0.f, tss = 0.f;
    #pragma unroll
    for (int w = 0; w < 8; ++w) { ts += sh[w][0]; tss += sh[w][1]; }
    float mean = ts * (1.0f / Cn);
    float var  = tss * (1.0f / Cn) - mean * mean;
    float rs   = rsqrtf(var + 1e-5f);
    float y0 = (x0 - mean) * rs * g[tid]       + bb[tid];
    float y1 = (x1 - mean) * rs * g[tid + 256] + bb[tid + 256];
    if (outb) { outb[(size_t)row * Cn + tid] = (bf16_t)y0; outb[(size_t)row * Cn + tid + 256] = (bf16_t)y1; }
    if (outf) { outf[(size_t)row * Cn + tid] = y0;         outf[(size_t)row * Cn + tid + 256] = y1; }
}

// ---------------- depthwise conv + gelu ----------------
__global__ void dsa_depthwise(const float* __restrict__ z, const float* __restrict__ dw,
                              const float* __restrict__ db, bf16_t* __restrict__ zg) {
    int idx = blockIdx.x * 256 + threadIdx.x;
    if (idx >= Bn * Tn * Cn) return;
    int c = idx & (Cn - 1);
    int t = (idx >> 9) & (Tn - 1);
    int b = idx >> 20;
    float acc = db[c];
    #pragma unroll
    for (int k = 0; k < 3; ++k) {
        int tt = t + k - 1;
        if (tt >= 0 && tt < Tn)
            acc += dw[c * 3 + k] * z[((size_t)b * Tn + tt) * Cn + c];
    }
    zg[idx] = (bf16_t)gelu_exact(acc);
}

// ---------------- TDM double-buffered WMMA GEMM with epilogues ----------------
// C[M,N] (f32 accum) = A[M,K](bf16, row stride lda) @ Bt[K,N](bf16 row-major)
// Block: 256 threads = 8 waves, tile 128(M) x 64(N); K staged in 64-steps via TDM.
#define EPI_CONV    1
#define EPI_QKV     2
#define EPI_ATTNOUT 3
#define EPI_MLP1    4
#define EPI_MLP2    5
#define EPI_FINAL   6

__global__ __launch_bounds__(256) void gemm_wmma(
    const bf16_t* __restrict__ A, int lda, long long strideA,
    const bf16_t* __restrict__ Bt, const float* __restrict__ bias,
    int M, int N, int K, int epi,
    const float* __restrict__ aux,
    float* __restrict__ outf,
    bf16_t* __restrict__ ob0, bf16_t* __restrict__ ob1, bf16_t* __restrict__ ob2,
    const float* __restrict__ sA, const float* __restrict__ sB) {

    __shared__ __align__(16) bf16_t tA[2][128 * 64];   // 2 x 16KB
    __shared__ __align__(16) bf16_t tB[2][64 * 64];    // 2 x 8KB

    int tid  = threadIdx.x;
    int lane = tid & 31;
    int wave = tid >> 5;
    int l16  = lane & 15;
    bool hiL = lane >= 16;
    int koff = hiL ? 8 : 0;

    int wm = wave >> 1;          // 0..3
    int wn = wave & 1;           // 0..1
    int z = blockIdx.z;

    const bf16_t* Abase = A + (size_t)z * strideA + (size_t)blockIdx.y * 128 * lda;
    const bf16_t* Bbase = Bt + (size_t)blockIdx.x * 64;

    v8f acc00 = {}, acc01 = {}, acc10 = {}, acc11 = {};

    int S = K >> 6;  // K / 64
    if (wave == 0) {
        tdm_load_2d((unsigned)(size_t)&tA[0][0], Abase, 64, 128, (unsigned)lda);
        tdm_load_2d((unsigned)(size_t)&tB[0][0], Bbase, 64, 64, (unsigned)N);
    }

    for (int s = 0; s < S; ++s) {
        int cur = s & 1;
        if (wave == 0) __builtin_amdgcn_s_wait_tensorcnt(0); // buf[cur] landed
        __syncthreads();                                     // publish; prior compute done
        if (wave == 0 && (s + 1) < S) {
            int nxt = (s + 1) & 1;
            tdm_load_2d((unsigned)(size_t)&tA[nxt][0], Abase + (size_t)(s + 1) * 64, 64, 128, (unsigned)lda);
            tdm_load_2d((unsigned)(size_t)&tB[nxt][0], Bbase + (size_t)(s + 1) * 64 * N, 64, 64, (unsigned)N);
        }
        #pragma unroll
        for (int kk = 0; kk < 2; ++kk) {       // two 32-K substeps per staged tile
            Frag a0, a1, b0, b1;
            const bf16_t* la = &tA[cur][(size_t)(wm * 32 + l16) * 64 + kk * 32 + koff];
            a0.f4[0] = *(const float4*)la;
            a0.f4[1] = *(const float4*)(la + 16);
            const bf16_t* la1 = la + 16 * 64;
            a1.f4[0] = *(const float4*)la1;
            a1.f4[1] = *(const float4*)(la1 + 16);
            const bf16_t* lb = &tB[cur][(size_t)(kk * 32 + koff) * 64 + wn * 32 + l16];
            #pragma unroll
            for (int e = 0; e < 16; ++e) {
                int kr = (e < 8) ? e : e + 8;
                b0.v[e] = lb[kr * 64];
                b1.v[e] = lb[kr * 64 + 16];
            }
            acc00 = wmma_bf16(a0, b0, acc00);
            acc01 = wmma_bf16(a0, b1, acc01);
            acc10 = wmma_bf16(a1, b0, acc10);
            acc11 = wmma_bf16(a1, b1, acc11);
        }
        __syncthreads();                                     // done reading buf[cur]
    }

    int rowBase = blockIdx.y * 128 + wm * 32;
    int colBase = blockIdx.x * 64 + wn * 32;
    float sa = (epi == EPI_FINAL) ? sA[0] : 0.0f;
    float sb = (epi == EPI_FINAL) ? sB[0] : 0.0f;

    #pragma unroll
    for (int mi = 0; mi < 2; ++mi) {
        #pragma unroll
        for (int ni = 0; ni < 2; ++ni) {
            v8f accv = (mi == 0) ? (ni == 0 ? acc00 : acc01) : (ni == 0 ? acc10 : acc11);
            int col = colBase + ni * 16 + l16;
            float bv = bias ? bias[col] : 0.0f;
            #pragma unroll
            for (int r = 0; r < 8; ++r) {
                int rowl = rowBase + mi * 16 + r + (hiL ? 8 : 0);
                size_t gRow = (size_t)z * M + rowl;
                float val = accv[r] + bv;
                if (epi == EPI_CONV) {
                    size_t o = gRow * Cn + col;
                    outf[o] = aux[o] + gelu_exact(val);
                } else if (epi == EPI_QKV) {
                    int b  = (int)(gRow >> 11);
                    int t  = (int)(gRow & (Tn - 1));
                    int sec = col >> 9;
                    int ch  = col & (Cn - 1);
                    int h   = ch >> 6;
                    int hd  = ch & (HDn - 1);
                    size_t bh = (size_t)b * Hn + h;
                    if (sec == 0)      ob0[(bh * Tn + t) * HDn + hd] = (bf16_t)val;
                    else if (sec == 1) ob1[(bh * HDn + hd) * Tn + t] = (bf16_t)val; // k transposed
                    else               ob2[(bh * Tn + t) * HDn + hd] = (bf16_t)val;
                } else if (epi == EPI_ATTNOUT) {
                    size_t o = gRow * Cn + col;
                    outf[o] = val + aux[o];
                } else if (epi == EPI_MLP1) {
                    ob0[gRow * (size_t)N + col] = (bf16_t)gelu_exact(val);
                } else if (epi == EPI_MLP2) {
                    outf[gRow * Cn + col] = val;
                } else { // EPI_FINAL
                    size_t o = gRow * Cn + col;
                    outf[o] = sa * aux[o] + sb * val;
                }
            }
        }
    }
}

// ---------------- flash attention ----------------
__global__ __launch_bounds__(128) void attn_flash(
    const bf16_t* __restrict__ q,   // [B,H,T,HD]
    const bf16_t* __restrict__ kT,  // [B,H,HD,T]
    const bf16_t* __restrict__ v,   // [B,H,T,HD]
    const float* __restrict__ Aarr, // [B,T]
    const float* __restrict__ alpha_bias,
    bf16_t* __restrict__ attnb) {   // [B,T,C]

    __shared__ __align__(16) bf16_t pbuf[4][16][32];

    int lane = threadIdx.x & 31;
    int wave = threadIdx.x >> 5;
    int l16  = lane & 15;
    bool hiL = lane >= 16;
    int koff = hiL ? 8 : 0;
    int rofs = hiL ? 8 : 0;

    int gw = blockIdx.x * 4 + wave;
    int bh = gw >> 7;            // T/16 = 128 q-tiles per (b,h)
    int qt = gw & 127;
    int b  = bh >> 3;            // H = 8
    int h  = bh & 7;

    const bf16_t* qp = q + ((size_t)bh * Tn + qt * 16 + l16) * HDn + koff;
    Frag qf0, qf1;
    qf0.f4[0] = *(const float4*)qp;
    qf0.f4[1] = *(const float4*)(qp + 16);
    qf1.f4[0] = *(const float4*)(qp + 32);
    qf1.f4[1] = *(const float4*)(qp + 48);

    float alpha = alpha_bias[0];
    float rowA[8];
    #pragma unroll
    for (int r = 0; r < 8; ++r)
        rowA[r] = Aarr[(size_t)b * Tn + qt * 16 + rofs + r] * alpha;

    v8f o0 = {}, o1 = {}, o2 = {}, o3 = {};
    float m[8], l[8];
    #pragma unroll
    for (int r = 0; r < 8; ++r) { m[r] = -3.0e38f; l[r] = 0.0f; }

    const bf16_t* kbase = kT + (size_t)bh * HDn * Tn;
    const bf16_t* vbase = v + (size_t)bh * Tn * HDn;

    for (int kb = 0; kb < Tn / 32; ++kb) {
        int n0 = kb * 32;
        v8f s0 = {}, s1 = {};
        const bf16_t* kp = kbase + (size_t)koff * Tn + n0 + l16;
        Frag kf;
        #pragma unroll
        for (int e = 0; e < 16; ++e) kf.v[e] = kp[(size_t)((e < 8) ? e : e + 8) * Tn];
        s0 = wmma_bf16(qf0, kf, s0);
        #pragma unroll
        for (int e = 0; e < 16; ++e) kf.v[e] = kp[(size_t)((e < 8) ? e : e + 8) * Tn + 16];
        s1 = wmma_bf16(qf0, kf, s1);
        const bf16_t* kp2 = kp + (size_t)32 * Tn;
        #pragma unroll
        for (int e = 0; e < 16; ++e) kf.v[e] = kp2[(size_t)((e < 8) ? e : e + 8) * Tn];
        s0 = wmma_bf16(qf1, kf, s0);
        #pragma unroll
        for (int e = 0; e < 16; ++e) kf.v[e] = kp2[(size_t)((e < 8) ? e : e + 8) * Tn + 16];
        s1 = wmma_bf16(qf1, kf, s1);

        float Aj0 = Aarr[(size_t)b * Tn + n0 + l16];
        float Aj1 = Aarr[(size_t)b * Tn + n0 + 16 + l16];

        float sc[8];
        #pragma unroll
        for (int r = 0; r < 8; ++r) {
            float x0 = s0[r] * 0.125f + rowA[r] * Aj0;
            float x1 = s1[r] * 0.125f + rowA[r] * Aj1;
            float mx = fmaxf(x0, x1);
            #pragma unroll
            for (int msk = 8; msk >= 1; msk >>= 1)
                mx = fmaxf(mx, __shfl_xor(mx, msk, 32));
            float mn = fmaxf(m[r], mx);
            sc[r] = expf(m[r] - mn);
            float p0 = expf(x0 - mn);
            float p1 = expf(x1 - mn);
            float ps = p0 + p1;
            #pragma unroll
            for (int msk = 8; msk >= 1; msk >>= 1)
                ps += __shfl_xor(ps, msk, 32);
            l[r] = l[r] * sc[r] + ps;
            m[r] = mn;
            pbuf[wave][rofs + r][l16]      = (bf16_t)p0;
            pbuf[wave][rofs + r][16 + l16] = (bf16_t)p1;
        }
        #pragma unroll
        for (int r = 0; r < 8; ++r) {
            o0[r] *= sc[r]; o1[r] *= sc[r]; o2[r] *= sc[r]; o3[r] *= sc[r];
        }

        asm volatile("s_wait_dscnt 0" ::: "memory");   // same-wave LDS RAW

        Frag pf;
        const bf16_t* pp = &pbuf[wave][l16][koff];
        pf.f4[0] = *(const float4*)pp;
        pf.f4[1] = *(const float4*)(pp + 16);

        const bf16_t* vp = vbase + (size_t)(n0 + koff) * HDn + l16;
        Frag vf;
        #pragma unroll
        for (int e = 0; e < 16; ++e) vf.v[e] = vp[((e < 8) ? e : e + 8) * HDn + 0];
        o0 = wmma_bf16(pf, vf, o0);
        #pragma unroll
        for (int e = 0; e < 16; ++e) vf.v[e] = vp[((e < 8) ? e : e + 8) * HDn + 16];
        o1 = wmma_bf16(pf, vf, o1);
        #pragma unroll
        for (int e = 0; e < 16; ++e) vf.v[e] = vp[((e < 8) ? e : e + 8) * HDn + 32];
        o2 = wmma_bf16(pf, vf, o2);
        #pragma unroll
        for (int e = 0; e < 16; ++e) vf.v[e] = vp[((e < 8) ? e : e + 8) * HDn + 48];
        o3 = wmma_bf16(pf, vf, o3);

        asm volatile("" ::: "memory"); // keep next-iter LDS stores after these reads
    }

    #pragma unroll
    for (int r = 0; r < 8; ++r) {
        int trow = qt * 16 + rofs + r;
        size_t base = ((size_t)b * Tn + trow) * Cn + h * HDn + l16;
        float inv = 1.0f / l[r];
        attnb[base + 0]  = (bf16_t)(o0[r] * inv);
        attnb[base + 16] = (bf16_t)(o1[r] * inv);
        attnb[base + 32] = (bf16_t)(o2[r] * inv);
        attnb[base + 48] = (bf16_t)(o3[r] * inv);
    }
}

// ---------------- host launch ----------------
extern "C" void kernel_launch(void* const* d_in, const int* in_sizes, int n_in,
                              void* d_out, int out_size, void* d_ws, size_t ws_size,
                              hipStream_t stream) {
    const float* x          = (const float*)d_in[0];
    const float* Aarr       = (const float*)d_in[1];
    const float* alpha_bias = (const float*)d_in[2];
    const float* dst_alpha  = (const float*)d_in[3];
    const float* dst_beta   = (const float*)d_in[4];
    const float* conv1_w    = (const float*)d_in[5];
    const float* conv1_b    = (const float*)d_in[6];
    const float* ln1_g      = (const float*)d_in[7];
    const float* ln1_b      = (const float*)d_in[8];
    const float* in_proj_w  = (const float*)d_in[9];
    const float* in_proj_b  = (const float*)d_in[10];
    const float* out_w      = (const float*)d_in[11];
    const float* out_b      = (const float*)d_in[12];
    const float* ln2_g      = (const float*)d_in[13];
    const float* ln2_b      = (const float*)d_in[14];
    const float* mlp_w1     = (const float*)d_in[15];
    const float* mlp_b1     = (const float*)d_in[16];
    const float* mlp_w2     = (const float*)d_in[17];
    const float* mlp_b2     = (const float*)d_in[18];
    const float* dsa_ln_g   = (const float*)d_in[19];
    const float* dsa_ln_b   = (const float*)d_in[20];
    const float* dsa_dw     = (const float*)d_in[21];
    const float* dsa_db     = (const float*)d_in[22];
    const float* dsa_pw     = (const float*)d_in[23];
    const float* dsa_pb     = (const float*)d_in[24];

    char* ws = (char*)d_ws;
    size_t off = 0;
    auto alloc = [&](size_t bytes) { size_t o = off; off += (bytes + 255) & ~(size_t)255; return o; };

    const size_t M = (size_t)Bn * Tn;                     // 8192
    bf16_t* wc    = (bf16_t*)(ws + alloc(3 * Cn * Cn * 2));
    bf16_t* wqkv  = (bf16_t*)(ws + alloc((size_t)Cn * 3 * Cn * 2));
    bf16_t* wout  = (bf16_t*)(ws + alloc((size_t)Cn * Cn * 2));
    bf16_t* w1    = (bf16_t*)(ws + alloc((size_t)Cn * FFn * 2));
    bf16_t* w2    = (bf16_t*)(ws + alloc((size_t)FFn * Cn * 2));
    bf16_t* wpw   = (bf16_t*)(ws + alloc((size_t)Cn * Cn * 2));
    bf16_t* xpad  = (bf16_t*)(ws + alloc((size_t)Bn * (Tn + 2) * Cn * 2));
    float*  F1    = (float*)(ws + alloc(M * Cn * 4));
    float*  F2    = (float*)(ws + alloc(M * Cn * 4));
    bf16_t* hbuf  = (bf16_t*)(ws + alloc(M * Cn * 2));
    bf16_t* qbuf  = (bf16_t*)(ws + alloc(M * Cn * 2));
    bf16_t* ktbuf = (bf16_t*)(ws + alloc(M * Cn * 2));
    bf16_t* vbuf  = (bf16_t*)(ws + alloc(M * Cn * 2));
    bf16_t* attnb = (bf16_t*)(ws + alloc(M * Cn * 2));
    bf16_t* mhbuf = (bf16_t*)(ws + alloc(M * FFn * 2));

    float* out = (float*)d_out;

    pack_conv_w<<<(3 * Cn * Cn + 255) / 256, 256, 0, stream>>>(conv1_w, wc);
    pack_transpose<<<(3 * Cn * Cn + 255) / 256, 256, 0, stream>>>(in_proj_w, wqkv, 3 * Cn, Cn);
    pack_transpose<<<(Cn * Cn + 255) / 256, 256, 0, stream>>>(out_w, wout, Cn, Cn);
    pack_transpose<<<(FFn * Cn + 255) / 256, 256, 0, stream>>>(mlp_w1, w1, FFn, Cn);
    pack_transpose<<<(Cn * FFn + 255) / 256, 256, 0, stream>>>(mlp_w2, w2, Cn, FFn);
    pack_transpose<<<(Cn * Cn + 255) / 256, 256, 0, stream>>>(dsa_pw, wpw, Cn, Cn);
    pack_xpad<<<(Bn * (Tn + 2) * Cn + 255) / 256, 256, 0, stream>>>(x, xpad);

    // conv1 as GEMM (per batch: M=T, K=3C, N=C, overlapping rows, lda=C)
    gemm_wmma<<<dim3(Cn / 64, Tn / 128, Bn), 256, 0, stream>>>(
        xpad, Cn, (long long)(Tn + 2) * Cn, wc, conv1_b,
        Tn, Cn, 3 * Cn, EPI_CONV, x, F1, nullptr, nullptr, nullptr, nullptr, nullptr);

    layernorm_k<<<(int)M, 256, 0, stream>>>(F1, ln1_g, ln1_b, hbuf, nullptr);

    gemm_wmma<<<dim3(3 * Cn / 64, M / 128, 1), 256, 0, stream>>>(
        hbuf, Cn, 0, wqkv, in_proj_b,
        (int)M, 3 * Cn, Cn, EPI_QKV, nullptr, nullptr, qbuf, ktbuf, vbuf, nullptr, nullptr);

    attn_flash<<<(Bn * Hn * (Tn / 16)) / 4, 128, 0, stream>>>(
        qbuf, ktbuf, vbuf, Aarr, alpha_bias, attnb);

    gemm_wmma<<<dim3(Cn / 64, M / 128, 1), 256, 0, stream>>>(
        attnb, Cn, 0, wout, out_b,
        (int)M, Cn, Cn, EPI_ATTNOUT, x, F2, nullptr, nullptr, nullptr, nullptr, nullptr);

    layernorm_k<<<(int)M, 256, 0, stream>>>(F2, ln2_g, ln2_b, qbuf, nullptr);

    gemm_wmma<<<dim3(FFn / 64, M / 128, 1), 256, 0, stream>>>(
        qbuf, Cn, 0, w1, mlp_b1,
        (int)M, FFn, Cn, EPI_MLP1, nullptr, nullptr, mhbuf, nullptr, nullptr, nullptr, nullptr);

    gemm_wmma<<<dim3(Cn / 64, M / 128, 1), 256, 0, stream>>>(
        mhbuf, FFn, 0, w2, mlp_b2,
        (int)M, Cn, FFn, EPI_MLP2, nullptr, F2, nullptr, nullptr, nullptr, nullptr, nullptr);

    layernorm_k<<<(int)M, 256, 0, stream>>>(x, dsa_ln_g, dsa_ln_b, nullptr, F1);

    dsa_depthwise<<<(int)(M * Cn + 255) / 256, 256, 0, stream>>>(F1, dsa_dw, dsa_db, hbuf);

    gemm_wmma<<<dim3(Cn / 64, M / 128, 1), 256, 0, stream>>>(
        hbuf, Cn, 0, wpw, dsa_pb,
        (int)M, Cn, Cn, EPI_FINAL, F2, out, nullptr, nullptr, nullptr, dst_alpha, dst_beta);
}
